// LinearAttention_43654047596635
// MI455X (gfx1250) — matrix operand order repaired and checked by reading, not compiled, not run on hardware
//
#include <hip/hip_runtime.h>
#include <hip/hip_bf16.h>
#include <math.h>

// Problem constants (from reference)
#define Hh 4
#define Tt 1024
#define Dd 8192

typedef __bf16 bf16_t;
typedef __attribute__((ext_vector_type(2)))  bf16_t v2bf;
typedef __attribute__((ext_vector_type(8)))  bf16_t v8bf;
typedef __attribute__((ext_vector_type(16))) bf16_t v16bf;
typedef __attribute__((ext_vector_type(8)))  float  v8f;

typedef __attribute__((ext_vector_type(4))) unsigned int u32x4;
typedef __attribute__((ext_vector_type(4))) int          i32x4;
typedef __attribute__((ext_vector_type(8))) int          i32x8;

// GEMM tiling
#define BM 128
#define BN 128
#define BK 64
#define LSTR 80   // padded LDS row stride (bf16): 128B data + 32B pad = 160B rows

// ---------------------------------------------------------------------------
// TDM: DMA one 128-row x 64-elem bf16 2D tile (row stride = strideElems) from
// global memory into LDS, inserting 32B of padding after every 128B row so the
// LDS image has row stride LSTR(=80) bf16 elems.  D# layout per CDNA5 ISA
// ch.8 (group0: count/lds_addr/global_addr/type; group1: data_size=2B,
// pad_interval=4 (128B), pad_amount=7 (32B), dims/strides).  Tracked with
// TENSORcnt; in-order completion per wave.
// ---------------------------------------------------------------------------
__device__ __forceinline__ void tdm_load_tile(const void* gsrc, const void* lds,
                                              unsigned strideElems) {
  unsigned long long ga = (unsigned long long)(uintptr_t)gsrc;
  unsigned ldsOff = (unsigned)(uintptr_t)lds;   // low 32 bits of generic = LDS byte addr
  u32x4 g0;
  g0.x = 1u;                                           // count=1, user descriptor
  g0.y = ldsOff;                                       // lds_addr
  g0.z = (unsigned)(ga & 0xFFFFFFFFu);                 // global_addr[31:0]
  g0.w = (unsigned)((ga >> 32) & 0x01FFFFFFu)          // global_addr[56:32]
       | 0x80000000u;                                  // type = 2 ("image")
  i32x8 g1;
  g1[0] = (1 << 16)      // data_size = 1 -> 2 bytes
        | (1 << 20)      // pad_enable
        | (4 << 22)      // pad_interval: pad after 2^(4+3)=128 bytes
        | (7 << 25);     // pad_amount: (7+1)*4 = 32 bytes
  g1[1] = (int)((strideElems & 0xFFFFu) << 16);        // tensor_dim0[15:0]
  g1[2] = (int)((strideElems >> 16) | (128u << 16));   // tensor_dim0[31:16], tensor_dim1=128
  g1[3] = (int)(64u << 16);                            // tile_dim0 = 64 elems
  g1[4] = 128;                                         // tile_dim1 = 128, tile_dim2 = 0
  g1[5] = (int)strideElems;                            // tensor_dim0_stride[31:0]
  g1[6] = 0;                                           // stride[47:32]=0, dim1_stride=0
  g1[7] = 0;
  i32x4 z4 = {0, 0, 0, 0};
#if defined(__clang_major__) && (__clang_major__ >= 23)
  i32x8 z8 = {0, 0, 0, 0, 0, 0, 0, 0};
  __builtin_amdgcn_tensor_load_to_lds(g0, g1, z4, z4, z8, 0);
#else
  __builtin_amdgcn_tensor_load_to_lds(g0, g1, z4, z4, 0);
#endif
}

// ---------------------------------------------------------------------------
// Kernel 1: RoPE(query) fp32 -> bf16.  THETA = 2^16 so freq = 2^(-i/256)/(2pi).
// ---------------------------------------------------------------------------
__global__ __launch_bounds__(256) void rope_bf16_kernel(const float* __restrict__ q,
                                                        bf16_t* __restrict__ out) {
  const int HALF = Dd / 2;
  size_t idx = (size_t)blockIdx.x * blockDim.x + threadIdx.x;  // pair id over H*T*HALF
  int   i  = (int)(idx % HALF);
  size_t ht = idx / HALF;            // h*T + t
  int   t  = (int)(ht % Tt);

  float f = exp2f(-(float)i * (1.0f / 256.0f)) * 0.15915494309189535f;
  float p = (float)t * f;
  p = (p - floorf(p)) * 6.283185307179586f;
  float sp, cp;
  __sincosf(p, &sp, &cp);

  float2 v = ((const float2*)q)[ht * HALF + i];
  v2bf o;
  o.x = (bf16_t)(v.x * cp - v.y * sp);
  o.y = (bf16_t)(v.y * cp + v.x * sp);
  ((v2bf*)out)[ht * HALF + i] = o;
}

// ---------------------------------------------------------------------------
// Kernel 1b: transpose + convert V fp32 [h][t][d] -> Vt bf16 [h][d][t] so the
// second GEMM's B tiles become plain 2D bf16 tiles (TDM-loadable, no VALU
// conversion in the hot loop).  32x32 LDS-tiled transpose, 256 threads.
// ---------------------------------------------------------------------------
__global__ __launch_bounds__(256) void vt_kernel(const float* __restrict__ V,
                                                 bf16_t* __restrict__ Vt) {
  __shared__ float tile[32][33];
  const int h  = blockIdx.z;
  const int t0 = blockIdx.y * 32;
  const int d0 = blockIdx.x * 32;
  const int lx = threadIdx.x & 31;
  const int ly = threadIdx.x >> 5;      // 0..7
  for (int r = 0; r < 32; r += 8)
    tile[ly + r][lx] = V[((size_t)(h * Tt + t0 + ly + r)) * Dd + d0 + lx];
  __syncthreads();
  for (int r = 0; r < 32; r += 8)
    Vt[((size_t)(h * Dd + d0 + ly + r)) * Tt + t0 + lx] = (bf16_t)tile[lx][ly + r];
}

// ---------------------------------------------------------------------------
// Kernel 2: S[h,t,s] = (t > s) ? scale * dot(Qr[h,t,:], Qr[h,s,:]) : 0  (bf16)
// 128x128 block, 8 waves (2x4), wave tile 64x32.  K = D = 8192, BK = 64,
// TDM double-buffered LDS staging overlapped with WMMA compute.
// ---------------------------------------------------------------------------
__global__ __launch_bounds__(256) void scores_kernel(const bf16_t* __restrict__ Q,
                                                     bf16_t* __restrict__ S) {
  const int h  = blockIdx.z;
  const int tb = blockIdx.y * BM;
  const int sb = blockIdx.x * BN;
  const int tid = threadIdx.x;

  if (sb > tb) {  // strictly above the diagonal: all-zero block
    const uint4 z = make_uint4(0u, 0u, 0u, 0u);
    for (int j = 0; j < 8; ++j) {
      int c = tid + 256 * j;
      int row = c >> 4;
      int col = (c & 15) * 8;
      *(uint4*)(S + ((size_t)(h * Tt + tb + row)) * Tt + sb + col) = z;
    }
    return;
  }

  __shared__ bf16_t As[2 * BM * LSTR];
  __shared__ bf16_t Bs[2 * BN * LSTR];

  const int wave = tid >> 5;
  const int lane = tid & 31;
  const int wm = wave >> 2;
  const int wn = wave & 3;
  const int lo = lane & 15;
  const int hi = lane >> 4;

  v8f acc[4][2] = {};

  const bf16_t* Abase = Q + ((size_t)(h * Tt + tb)) * Dd;
  const bf16_t* Bbase = Q + ((size_t)(h * Tt + sb)) * Dd;
  const int nk = Dd / BK;

  if (tid < 32) {  // wave 0 drives the TDM pipeline
    tdm_load_tile(Abase, As, Dd);
    tdm_load_tile(Bbase, Bs, Dd);
  }

  for (int i = 0; i < nk; ++i) {
    const int cur = i & 1;
    if (tid < 32) {
      if (i + 1 < nk) {
        tdm_load_tile(Abase + (size_t)(i + 1) * BK, As + (cur ^ 1) * BM * LSTR, Dd);
        tdm_load_tile(Bbase + (size_t)(i + 1) * BK, Bs + (cur ^ 1) * BN * LSTR, Dd);
        __builtin_amdgcn_s_wait_tensorcnt(2);   // stage i retired (in-order)
      } else {
        __builtin_amdgcn_s_wait_tensorcnt(0);
      }
    }
    __syncthreads();

    const bf16_t* Ac = As + cur * BM * LSTR;
    const bf16_t* Bc = Bs + cur * BN * LSTR;
    for (int kk = 0; kk < BK / 32; ++kk) {
      v16bf af[4];
      v16bf bfm[2];
      for (int tm = 0; tm < 4; ++tm) {
        const bf16_t* p = Ac + (wm * 64 + tm * 16 + lo) * LSTR + kk * 32;
        v8bf* ap = (v8bf*)&af[tm];
        ap[0] = *(const v8bf*)(p + hi * 16);       // K 0-7 / 16-23
        ap[1] = *(const v8bf*)(p + hi * 16 + 8);   // K 8-15 / 24-31
      }
      for (int tn = 0; tn < 2; ++tn)
        bfm[tn] = *(const v16bf*)(Bc + (wn * 32 + tn * 16 + lo) * LSTR + kk * 32 + hi * 16);
      for (int tm = 0; tm < 4; ++tm)
        for (int tn = 0; tn < 2; ++tn)
          acc[tm][tn] = __builtin_amdgcn_wmma_f32_16x16x32_bf16(
              false, af[tm], false, bfm[tn], (short)0, acc[tm][tn], false, false);
    }
    __syncthreads();
  }

  const float scale = 0.011048543456039806f;   // 1/sqrt(8192)
  for (int tm = 0; tm < 4; ++tm) {
    for (int tn = 0; tn < 2; ++tn) {
      int s = sb + wn * 32 + tn * 16 + lo;
      for (int r = 0; r < 8; ++r) {
        int t = tb + wm * 64 + tm * 16 + hi * 8 + r;
        float val = acc[tm][tn][r] * scale;
        if (t <= s) val = 0.0f;
        S[((size_t)(h * Tt + t)) * Tt + s] = (bf16_t)val;
      }
    }
  }
}

// ---------------------------------------------------------------------------
// Kernel 3: out[h,t,d] = sum_s S[h,t,s] * Vt[h,d,s].  K truncated at the
// causal boundary.  Both operands are bf16 2D tiles -> pure TDM pipeline.
// ---------------------------------------------------------------------------
__global__ __launch_bounds__(256) void out_gemm_kernel(const bf16_t* __restrict__ S,
                                                       const bf16_t* __restrict__ Vt,
                                                       float* __restrict__ O) {
  const int h  = blockIdx.z;
  const int tb = blockIdx.y * BM;
  const int db = blockIdx.x * BN;
  const int tid = threadIdx.x;

  __shared__ bf16_t As[2 * BM * LSTR];   // [t][s] tiles
  __shared__ bf16_t Bs[2 * BN * LSTR];   // [d][s] tiles

  const int wave = tid >> 5;
  const int lane = tid & 31;
  const int wm = wave >> 2;
  const int wn = wave & 3;
  const int lo = lane & 15;
  const int hi = lane >> 4;

  v8f acc[4][2] = {};

  const bf16_t* Abase = S  + ((size_t)(h * Tt + tb)) * Tt;
  const bf16_t* Bbase = Vt + ((size_t)(h * Dd + db)) * Tt;
  const int nk = (tb + BM) / BK;         // causal: s < t

  if (tid < 32) {
    tdm_load_tile(Abase, As, Tt);
    tdm_load_tile(Bbase, Bs, Tt);
  }

  for (int i = 0; i < nk; ++i) {
    const int cur = i & 1;
    if (tid < 32) {
      if (i + 1 < nk) {
        tdm_load_tile(Abase + (size_t)(i + 1) * BK, As + (cur ^ 1) * BM * LSTR, Tt);
        tdm_load_tile(Bbase + (size_t)(i + 1) * BK, Bs + (cur ^ 1) * BN * LSTR, Tt);
        __builtin_amdgcn_s_wait_tensorcnt(2);
      } else {
        __builtin_amdgcn_s_wait_tensorcnt(0);
      }
    }
    __syncthreads();

    const bf16_t* Ac = As + cur * BM * LSTR;
    const bf16_t* Bc = Bs + cur * BN * LSTR;
    for (int kk = 0; kk < BK / 32; ++kk) {
      v16bf af[4];
      v16bf bfm[2];
      for (int tm = 0; tm < 4; ++tm) {
        const bf16_t* p = Ac + (wm * 64 + tm * 16 + lo) * LSTR + kk * 32;
        v8bf* ap = (v8bf*)&af[tm];
        ap[0] = *(const v8bf*)(p + hi * 16);
        ap[1] = *(const v8bf*)(p + hi * 16 + 8);
      }
      for (int tn = 0; tn < 2; ++tn)
        bfm[tn] = *(const v16bf*)(Bc + (wn * 32 + tn * 16 + lo) * LSTR + kk * 32 + hi * 16);
      for (int tm = 0; tm < 4; ++tm)
        for (int tn = 0; tn < 2; ++tn)
          acc[tm][tn] = __builtin_amdgcn_wmma_f32_16x16x32_bf16(
              false, af[tm], false, bfm[tn], (short)0, acc[tm][tn], false, false);
    }
    __syncthreads();
  }

  for (int tm = 0; tm < 4; ++tm) {
    for (int tn = 0; tn < 2; ++tn) {
      int d = db + wn * 32 + tn * 16 + lo;
      for (int r = 0; r < 8; ++r) {
        int t = tb + wm * 64 + tm * 16 + hi * 8 + r;
        O[((size_t)(h * Tt + t)) * Dd + d] = acc[tm][tn][r];
      }
    }
  }
}

// ---------------------------------------------------------------------------
// Launch.  ws layout:
//   [ropeQ bf16 : H*T*D*2 = 64MB][Vt bf16 : H*D*T*2 = 64MB][S bf16 : 8MB]
// ---------------------------------------------------------------------------
extern "C" void kernel_launch(void* const* d_in, const int* in_sizes, int n_in,
                              void* d_out, int out_size, void* d_ws, size_t ws_size,
                              hipStream_t stream) {
  (void)in_sizes; (void)n_in; (void)out_size; (void)ws_size;
  const float* q = (const float*)d_in[0];
  const float* v = (const float*)d_in[1];
  float* out = (float*)d_out;

  const size_t qBytes = (size_t)Hh * Tt * Dd * sizeof(bf16_t);
  bf16_t* ropeQ = (bf16_t*)d_ws;
  bf16_t* Vt    = (bf16_t*)((char*)d_ws + qBytes);
  bf16_t* Smat  = (bf16_t*)((char*)d_ws + 2 * qBytes);

  {  // 1) RoPE + bf16 convert
    size_t pairs = (size_t)Hh * Tt * (Dd / 2);
    rope_bf16_kernel<<<dim3((unsigned)(pairs / 256)), 256, 0, stream>>>(q, ropeQ);
  }
  {  // 1b) V transpose + bf16 convert
    dim3 grid(Dd / 32, Tt / 32, Hh);
    vt_kernel<<<grid, 256, 0, stream>>>(v, Vt);
  }
  {  // 2) masked scores GEMM
    dim3 grid(Tt / BN, Tt / BM, Hh);
    scores_kernel<<<grid, 256, 0, stream>>>(ropeQ, Smat);
  }
  {  // 3) output GEMM
    dim3 grid(Dd / BN, Tt / BM, Hh);
    out_gemm_kernel<<<grid, 256, 0, stream>>>(Smat, Vt, out);
  }
}
